// Bidirectional_LSTM_Attention_82489141887421
// MI455X (gfx1250) — matrix operand order repaired and checked by reading, not compiled, not run on hardware
//
#include <hip/hip_runtime.h>
#include <hip/hip_bf16.h>
#include <stdint.h>

#define VSZ   50000
#define ESZ   300
#define EP    320      // E padded to multiple of 32 (bf16 WMMA K)
#define HSZ   256
#define TSZ   512
#define BSZ   128
#define OUTC  3
#define ATTD  512      // 2H
#define G4    1024     // 4H
#define MTILES 4096    // (T*B)/16
#define NTILES 64      // G4/16

typedef float  v8f   __attribute__((ext_vector_type(8)));
typedef __bf16 v16bf __attribute__((ext_vector_type(16)));

union AFrag { v16bf v; uint4 q[2]; };
union CFrag { v8f v; float4 f4[2]; float f[8]; };

__device__ __forceinline__ unsigned short f2bf(float x) {
    union { float f; unsigned u; } c; c.f = x;
    unsigned r = c.u + 0x7FFFu + ((c.u >> 16) & 1u);   // RNE
    return (unsigned short)(r >> 16);
}
__device__ __forceinline__ float sigf(float x) { return 1.0f / (1.0f + __expf(-x)); }

// ---------------- denom[b] = sum_t tfidf[b][t] ----------------
__global__ __launch_bounds__(256) void k_denom(const float* __restrict__ tfidf,
                                               float* __restrict__ denom) {
    int b = blockIdx.x, tid = threadIdx.x;
    __shared__ float red[256];
    red[tid] = tfidf[b * TSZ + tid] + tfidf[b * TSZ + tid + 256];
    __syncthreads();
    for (int st = 128; st > 0; st >>= 1) {
        if (tid < st) red[tid] += red[tid + st];
        __syncthreads();
    }
    if (tid == 0) denom[b] = red[0];
}

// --------- emb[t*B+b][e] = bf16(emb_table[x[b][t]][e] * tfidf/denom), zero-padded ----------
__global__ __launch_bounds__(EP) void k_embed(const int* __restrict__ x,
                                              const float* __restrict__ tfidf,
                                              const float* __restrict__ denom,
                                              const float* __restrict__ embt,
                                              unsigned short* __restrict__ emb) {
    int row = blockIdx.x;            // t*B + b   (time-major)
    int t = row >> 7, b = row & 127;
    int e = threadIdx.x;
    int xi = x[b * TSZ + t];
    float scale = tfidf[b * TSZ + t] / denom[b];
    float v = (e < ESZ) ? embt[(size_t)xi * ESZ + e] * scale : 0.0f;
    emb[(size_t)row * EP + e] = f2bf(v);
}

// --------- pack f32 [rows][K] -> bf16 [rows][Kp], zero-padded ----------
__global__ void k_pack(const float* __restrict__ src, unsigned short* __restrict__ dst,
                       int K, int Kp) {
    int row = blockIdx.x, k = threadIdx.x;
    if (k < Kp) {
        float v = (k < K) ? src[(size_t)row * K + k] : 0.0f;
        dst[(size_t)row * Kp + k] = f2bf(v);
    }
}

// --------- big input GEMM: gx = emb @ Wih^T + (bih+bhh), stored in C-fragment layout ----------
__global__ __launch_bounds__(256) void k_ingemm(const unsigned short* __restrict__ emb, // [T*B][EP]
                                                const unsigned short* __restrict__ wih, // [G4][EP]
                                                const float* __restrict__ bih,
                                                const float* __restrict__ bhh,
                                                float* __restrict__ gx) {               // [MTILES][NTILES][32][8]
    int wave = threadIdx.x >> 5, lane = threadIdx.x & 31;
    int tile = blockIdx.x * 8 + wave;
    int mt = tile >> 6, nt = tile & 63;
    int l15 = lane & 15, hi = lane >> 4;
    int n = nt * 16 + l15;
    float bias = bih[n] + bhh[n];
    CFrag c;
#pragma unroll
    for (int r = 0; r < 8; ++r) c.f[r] = bias;

    int arow = mt * 16 + l15;
    const unsigned short* abase = emb + (size_t)arow * EP + hi * 8;   // A: K = hi*8 + {0..7,16..23}
    const unsigned short* bbase = wih + (size_t)n * EP + hi * 8;      // B: K = hi*8 + {0..7,16..23}
#pragma unroll
    for (int kt = 0; kt < EP / 32; ++kt) {
        AFrag a, b;
        a.q[0] = *(const uint4*)(abase + kt * 32);
        a.q[1] = *(const uint4*)(abase + kt * 32 + 16);
        b.q[0] = *(const uint4*)(bbase + kt * 32);
        b.q[1] = *(const uint4*)(bbase + kt * 32 + 16);
        c.v = __builtin_amdgcn_wmma_f32_16x16x32_bf16(false, a.v, false, b.v,
                                                      (short)0, c.v, false, false);
    }
    float4* dst = (float4*)(gx + (((size_t)mt * NTILES + nt) * 32 + lane) * 8);
    dst[0] = c.f4[0];
    dst[1] = c.f4[1];
}

// --------- recurrent LSTM: one block per direction, h in LDS, c/len in registers ----------
__global__ __launch_bounds__(1024) void k_lstm(const float* __restrict__ gx_f,
                                               const float* __restrict__ gx_b,
                                               const unsigned short* __restrict__ whh_f,
                                               const unsigned short* __restrict__ whh_b,
                                               const int* __restrict__ lengths,
                                               float* __restrict__ ht) {   // [B][T][ATTD]
    int dir = blockIdx.x;
    const float* gx = dir ? gx_b : gx_f;
    const unsigned short* whh = dir ? whh_b : whh_f;

    __shared__ unsigned short hsh[BSZ * HSZ];          // 64 KB: h_{t-1} in bf16
    {
        unsigned* p = (unsigned*)hsh;
        for (int i = threadIdx.x; i < BSZ * HSZ / 2; i += 1024) p[i] = 0u;
    }
    int wave = threadIdx.x >> 5, lane = threadIdx.x & 31;
    int wm = wave >> 4, wn = wave & 15;                // wm: M half (64 rows), wn: j tile
    int l15 = lane & 15, hi = lane >> 4;
    int j = wn * 16 + l15;

    float creg[4][8];
    int   lenr[4][8];
#pragma unroll
    for (int mt = 0; mt < 4; ++mt)
#pragma unroll
        for (int r = 0; r < 8; ++r) {
            creg[mt][r] = 0.0f;
            lenr[mt][r] = lengths[wm * 64 + mt * 16 + r + 8 * hi];
        }
    __syncthreads();

    for (int t = 0; t < TSZ; ++t) {
        int te = dir ? (TSZ - 1 - t) : t;
        float hnew[4][8];
#pragma unroll
        for (int mt = 0; mt < 4; ++mt) {
            int mtg = te * 8 + wm * 4 + mt;            // global M-tile: row = te*128 + b
            CFrag acc[4];
#pragma unroll
            for (int q = 0; q < 4; ++q) {              // init from precomputed x-gates (+bias)
                const float4* src = (const float4*)(gx +
                    (((size_t)mtg * NTILES + (q * 16 + wn)) * 32 + lane) * 8);
                acc[q].f4[0] = src[0];
                acc[q].f4[1] = src[1];
            }
            int arow = wm * 64 + mt * 16 + l15;
            const unsigned short* abase = hsh + arow * HSZ + hi * 8;
#pragma unroll
            for (int kt = 0; kt < HSZ / 32; ++kt) {
                AFrag a;
                a.q[0] = *(const uint4*)(abase + kt * 32);
                a.q[1] = *(const uint4*)(abase + kt * 32 + 16);
#pragma unroll
                for (int q = 0; q < 4; ++q) {
                    const unsigned short* bb = whh + (size_t)(q * 256 + j) * HSZ + kt * 32 + hi * 8;
                    AFrag b;
                    b.q[0] = *(const uint4*)(bb);
                    b.q[1] = *(const uint4*)(bb + 16);
                    acc[q].v = __builtin_amdgcn_wmma_f32_16x16x32_bf16(false, a.v, false, b.v,
                                                                      (short)0, acc[q].v, false, false);
                }
            }
#pragma unroll
            for (int r = 0; r < 8; ++r) {
                float iv = sigf(acc[0].f[r]);
                float fv = sigf(acc[1].f[r]);
                float gv = tanhf(acc[2].f[r]);
                float ov = sigf(acc[3].f[r]);
                float cn = fv * creg[mt][r] + iv * gv;
                float hn = ov * tanhf(cn);
                bool m = te < lenr[mt][r];
                if (m) creg[mt][r] = cn;
                hnew[mt][r] = hn;
                int b = wm * 64 + mt * 16 + r + 8 * hi;
                ht[((size_t)b * TSZ + te) * ATTD + dir * HSZ + j] = m ? hn : 0.0f;
            }
        }
        __syncthreads();                               // everyone done reading h_{t-1}
#pragma unroll
        for (int mt = 0; mt < 4; ++mt)
#pragma unroll
            for (int r = 0; r < 8; ++r) {
                if (te < lenr[mt][r]) {
                    int b = wm * 64 + mt * 16 + r + 8 * hi;
                    hsh[b * HSZ + j] = f2bf(hnew[mt][r]);
                }
            }
        __syncthreads();                               // h_t visible for next step
    }
}

// --------- attention + output head: one block per batch row ----------
__global__ __launch_bounds__(512) void k_att(const float* __restrict__ ht,
                                             const float* __restrict__ w_att,
                                             const float* __restrict__ b_att,
                                             const int* __restrict__ lengths,
                                             const float* __restrict__ W_out,
                                             const float* __restrict__ b_out,
                                             float* __restrict__ out) {
    int bidx = blockIdx.x, tid = threadIdx.x;
    __shared__ float watt[ATTD];
    __shared__ float sc[TSZ];
    __shared__ float red[512];
    __shared__ float rep[ATTD];
    watt[tid] = w_att[tid];
    __syncthreads();
    const float* hb = ht + (size_t)bidx * TSZ * ATTD;
    const float* row = hb + (size_t)tid * ATTD;
    float s = 0.0f;
    for (int d = 0; d < ATTD; ++d) s += row[d] * watt[d];
    s = tanhf(s + b_att[0]);
    sc[tid] = s;
    red[tid] = s;
    __syncthreads();
    for (int st = 256; st > 0; st >>= 1) {
        if (tid < st) red[tid] = fmaxf(red[tid], red[tid + st]);
        __syncthreads();
    }
    float mx = red[0];
    __syncthreads();
    int len = lengths[bidx];
    // softmax * mask, renormalized: unmasked denominator cancels exactly
    float e = (tid < len) ? __expf(sc[tid] - mx) : 0.0f;
    red[tid] = e;
    __syncthreads();
    for (int st = 256; st > 0; st >>= 1) {
        if (tid < st) red[tid] += red[tid + st];
        __syncthreads();
    }
    float z = red[0];
    __syncthreads();
    sc[tid] = e / z;                                   // att[t]
    __syncthreads();
    float acc = 0.0f;
    for (int t = 0; t < TSZ; ++t) acc += sc[t] * hb[(size_t)t * ATTD + tid];
    rep[tid] = acc;
    __syncthreads();
    if (tid < OUTC) {
        float o = b_out[tid];
        for (int d = 0; d < ATTD; ++d) o += W_out[tid * ATTD + d] * rep[d];
        out[bidx * OUTC + tid] = o;
    }
}

extern "C" void kernel_launch(void* const* d_in, const int* in_sizes, int n_in,
                              void* d_out, int out_size, void* d_ws, size_t ws_size,
                              hipStream_t stream) {
    const int*   x       = (const int*)d_in[0];
    const int*   lengths = (const int*)d_in[1];
    const float* tfidf   = (const float*)d_in[2];
    const float* embt    = (const float*)d_in[3];
    const float* Wih_f   = (const float*)d_in[4];
    const float* Whh_f   = (const float*)d_in[5];
    const float* bih_f   = (const float*)d_in[6];
    const float* bhh_f   = (const float*)d_in[7];
    const float* Wih_b   = (const float*)d_in[8];
    const float* Whh_b   = (const float*)d_in[9];
    const float* bih_b   = (const float*)d_in[10];
    const float* bhh_b   = (const float*)d_in[11];
    const float* w_att   = (const float*)d_in[12];
    const float* b_att   = (const float*)d_in[13];
    const float* W_out   = (const float*)d_in[14];
    const float* b_out   = (const float*)d_in[15];
    float* out = (float*)d_out;

    char* ws = (char*)d_ws;
    size_t off = 0;
    auto alloc = [&](size_t bytes) -> char* {
        char* p = ws + off;
        off += (bytes + 255) & ~(size_t)255;
        return p;
    };
    float*          denom = (float*)alloc((size_t)BSZ * 4);
    unsigned short* emb   = (unsigned short*)alloc((size_t)TSZ * BSZ * EP * 2);
    unsigned short* wihf  = (unsigned short*)alloc((size_t)G4 * EP * 2);
    unsigned short* wihb  = (unsigned short*)alloc((size_t)G4 * EP * 2);
    unsigned short* whhf  = (unsigned short*)alloc((size_t)G4 * HSZ * 2);
    unsigned short* whhb  = (unsigned short*)alloc((size_t)G4 * HSZ * 2);
    float*          gxf   = (float*)alloc((size_t)MTILES * NTILES * 256 * 4);
    float*          gxb   = (float*)alloc((size_t)MTILES * NTILES * 256 * 4);
    float*          ht    = (float*)alloc((size_t)BSZ * TSZ * ATTD * 4);

    k_denom<<<BSZ, 256, 0, stream>>>(tfidf, denom);
    k_embed<<<TSZ * BSZ, EP, 0, stream>>>(x, tfidf, denom, embt, emb);
    k_pack<<<G4, EP, 0, stream>>>(Wih_f, wihf, ESZ, EP);
    k_pack<<<G4, EP, 0, stream>>>(Wih_b, wihb, ESZ, EP);
    k_pack<<<G4, HSZ, 0, stream>>>(Whh_f, whhf, HSZ, HSZ);
    k_pack<<<G4, HSZ, 0, stream>>>(Whh_b, whhb, HSZ, HSZ);
    k_ingemm<<<MTILES * NTILES / 8, 256, 0, stream>>>(emb, wihf, bih_f, bhh_f, gxf);
    k_ingemm<<<MTILES * NTILES / 8, 256, 0, stream>>>(emb, wihb, bih_b, bhh_b, gxb);
    k_lstm<<<2, 1024, 0, stream>>>(gxf, gxb, whhf, whhb, lengths, ht);
    k_att<<<BSZ, 512, 0, stream>>>(ht, w_att, b_att, lengths, W_out, b_out, out);
}
